// cere_model_74955769250573
// MI455X (gfx1250) — matrix operand order repaired and checked by reading, not compiled, not run on hardware
//
#include <hip/hip_runtime.h>

typedef __bf16 bf16_t;
typedef __bf16 v16bf __attribute__((ext_vector_type(16)));
typedef __bf16 v8bf  __attribute__((ext_vector_type(8)));
typedef float  v8f   __attribute__((ext_vector_type(8)));

namespace {
constexpr int   BATCH  = 128;
constexpr int   TSTEPS = 19;
constexpr float THRESH = 0.5f;
constexpr float DECAY  = 0.2f;
constexpr int   NMF  = 1024;
constexpr int   NGC  = 2048;
constexpr int   NPC  = 1024;
constexpr int   NDCN = 1408;          // DCN*11
constexpr int   KM   = 320;           // NIN_MF*5
constexpr int   KU   = 384;           // NIN_MF*6
}

// Raw split workgroup barrier (no implied vmem waitcnt drain, unlike
// __syncthreads). Per-wave hazards are handled explicitly:
//  - s_wait_asynccnt before signaling (LDS tile chunk landed)
//  - backend inserts s_wait_dscnt before WMMA operand use
__device__ __forceinline__ void wg_barrier() {
  asm volatile("s_barrier_signal -1\n\t"
               "s_barrier_wait -1" ::: "memory");
}

// ---------------- utility kernels ----------------

__global__ void k_zero_u32(unsigned* __restrict__ p, size_t n) {
  size_t i = (size_t)blockIdx.x * blockDim.x + threadIdx.x;
  size_t stride = (size_t)gridDim.x * blockDim.x;
  for (; i < n; i += stride) p[i] = 0u;
}

__global__ void k_zero_scalar(float* p) { *p = 0.f; }

__global__ void k_f32_to_bf16(const float* __restrict__ s, bf16_t* __restrict__ d, size_t n) {
  size_t i = (size_t)blockIdx.x * blockDim.x + threadIdx.x;
  if (i < n) d[i] = (bf16_t)s[i];
}

// ---------------- fused GEMM + membrane update ----------------
// Block tile: M=128 (8 waves x 16 rows) x N=32 (2 column tiles per wave).
// K stepped by 64 -> 4 independent WMMA accumulation chains per wave.
// Software-pipelined: double-buffered LDS W tiles filled by
// global_load_async_to_lds_b128 one tile ahead (s_wait_asynccnt 1 ensures the
// current tile has landed while the next streams in), and A fragments
// register-double-buffered so their global_load_b128s fly under the WMMAs.
// All control flow is uniform (EXEC all-1s at every WMMA).
// Epilogue: mem = mem*DECAY*(1-spike_prev) + acc + bias ; spike = mem>THRESH.
struct AFrag {
  union { v16bf v; v8bf h[2]; } a0, a1;
};

__global__ __launch_bounds__(256)
void k_gemm_mem_spike(const bf16_t* __restrict__ X, int lda, int K,
                      const bf16_t* __restrict__ W, const float* __restrict__ bias,
                      float* __restrict__ mem, bf16_t* __restrict__ spike,
                      int N, float* spikeSum, float* accumOut)
{
  __shared__ __align__(128) bf16_t ldsW[2][32 * 64];  // 2 x 4 KB W tiles

  const int tid  = threadIdx.x;
  const int wave = tid >> 5;        // 0..7 -> M tile (rows 16*wave..)
  const int lane = tid & 31;
  const int nl   = lane & 15;       // column within 16-wide tile
  const int kh   = lane >> 4;       // K-half selector (WMMA A/B layout)
  const int n0   = blockIdx.x * 32; // 32-wide N slab

  // staging assignment: thread t copies one 16B chunk of the 4KB tile
  const int srow = tid >> 3;        // 0..31 : N row within tile
  const int sseg = tid & 7;         // 0..7  : 16B chunk within 64-K row
  const unsigned ldsDst0 =
      (unsigned)(size_t)(&ldsW[0][srow * 64 + sseg * 8]);  // LDS addr (low 32b)
  const unsigned ldsDst1 =
      (unsigned)(size_t)(&ldsW[1][srow * 64 + sseg * 8]);
  const bf16_t* gbase = W + (size_t)(n0 + srow) * K + sseg * 8;
  const bf16_t* xrow  = X + (size_t)(wave * 16 + nl) * lda;

  auto loadA = [&](int k) {
    AFrag f;
    f.a0.h[0] = *(const v8bf*)(xrow + k      + 16 * kh);
    f.a0.h[1] = *(const v8bf*)(xrow + k +  8 + 16 * kh);
    f.a1.h[0] = *(const v8bf*)(xrow + k + 32 + 16 * kh);
    f.a1.h[1] = *(const v8bf*)(xrow + k + 40 + 16 * kh);
    return f;
  };

  v8f acc00 = {0.f,0.f,0.f,0.f,0.f,0.f,0.f,0.f};
  v8f acc01 = acc00, acc10 = acc00, acc11 = acc00;

  // ---- pipeline prologue: tile 0 in flight, A(0) in flight ----
  asm volatile("global_load_async_to_lds_b128 %0, %1, off"
               :: "v"(ldsDst0), "v"(gbase) : "memory");
  AFrag aCur = loadA(0);

  const int nk = K >> 6;            // K / 64 tiles
  for (int i = 0; i < nk; ++i) {
    const int k0 = i << 6;
    const int kN = (i + 1 < nk) ? k0 + 64 : 0;   // last iter: harmless refetch

    wg_barrier();      // all waves done reading buffer (i+1)&1 (iter i-1) -> WAR safe

    // issue NEXT tile into the other LDS buffer; A(next) into registers
    {
      const unsigned dst = ((i + 1) & 1) ? ldsDst1 : ldsDst0;
      const bf16_t*  gsrc = gbase + kN;
      asm volatile("global_load_async_to_lds_b128 %0, %1, off"
                   :: "v"(dst), "v"(gsrc) : "memory");
      if (i + 2 < nk) __builtin_prefetch(gbase + k0 + 128, 0, 1);
    }
    AFrag aNxt = loadA(kN);

    // in-order async completion: <=1 outstanding => tile i has landed
    asm volatile("s_wait_asynccnt 1" ::: "memory");
    wg_barrier();      // everyone's chunk of tile i visible

    const bf16_t* lb = &ldsW[i & 1][0];
    // B fragments: 32 contiguous bytes of LDS row (ntile*16+nl), K-half kh
    const v16bf b00 = *(const v16bf*)&lb[(nl     ) * 64      + kh * 16];
    const v16bf b10 = *(const v16bf*)&lb[(16 + nl) * 64      + kh * 16];
    const v16bf b01 = *(const v16bf*)&lb[(nl     ) * 64 + 32 + kh * 16];
    const v16bf b11 = *(const v16bf*)&lb[(16 + nl) * 64 + 32 + kh * 16];

    acc00 = __builtin_amdgcn_wmma_f32_16x16x32_bf16(false, aCur.a0.v, false, b00,
                                                    (short)0, acc00, false, false);
    acc10 = __builtin_amdgcn_wmma_f32_16x16x32_bf16(false, aCur.a0.v, false, b10,
                                                    (short)0, acc10, false, false);
    acc01 = __builtin_amdgcn_wmma_f32_16x16x32_bf16(false, aCur.a1.v, false, b01,
                                                    (short)0, acc01, false, false);
    acc11 = __builtin_amdgcn_wmma_f32_16x16x32_bf16(false, aCur.a1.v, false, b11,
                                                    (short)0, acc11, false, false);
    aCur = aNxt;
  }
  const v8f acc0 = acc00 + acc01;   // column tile 0
  const v8f acc1 = acc10 + acc11;   // column tile 1

  // Epilogue: C/D layout — lane holds n = lane%16, M = v + 8*(lane>>4)
  const int   nA = n0 + nl;
  const int   nB = n0 + 16 + nl;
  const float biasA = bias[nA];
  const float biasB = bias[nB];
  float localSum = 0.f;
#pragma unroll
  for (int v = 0; v < 8; ++v) {
    const int r = wave * 16 + v + 8 * kh;
    {
      const size_t idx  = (size_t)r * N + nA;
      const float  val  = mem[idx] * DECAY * (1.f - (float)spike[idx]) + acc0[v] + biasA;
      mem[idx] = val;
      const float s = (val > THRESH) ? 1.f : 0.f;
      spike[idx] = (bf16_t)s;
      localSum += s;
      if (accumOut) accumOut[idx] += s;
    }
    {
      const size_t idx  = (size_t)r * N + nB;
      const float  val  = mem[idx] * DECAY * (1.f - (float)spike[idx]) + acc1[v] + biasB;
      mem[idx] = val;
      const float s = (val > THRESH) ? 1.f : 0.f;
      spike[idx] = (bf16_t)s;
      localSum += s;
      if (accumOut) accumOut[idx] += s;
    }
  }
  if (spikeSum) {
    // wave-level reduction -> one atomic per wave
    for (int o = 16; o > 0; o >>= 1) localSum += __shfl_down(localSum, o, 32);
    if (lane == 0) atomicAdd(spikeSum, localSum);
  }
}

// ---------------- torch-faithful interleaved gathers ----------------
// PF_in[b, j] = stacked[4b + j/2048, j%2048], stacked = [mGCs;uGCs;sGCs;fGCs]
__global__ void k_gather_pf(const bf16_t* __restrict__ s0, const bf16_t* __restrict__ s1,
                            const bf16_t* __restrict__ s2, const bf16_t* __restrict__ s3,
                            bf16_t* __restrict__ out)
{
  const int i = blockIdx.x * 256 + threadIdx.x;     // 128 * 8192 total
  if (i >= BATCH * 4 * NGC) return;
  const int b   = i >> 13;
  const int j   = i & 8191;
  const int seg = j >> 11;
  const int col = j & 2047;
  const int r   = 4 * b + seg;
  const bf16_t* src = (r < 128) ? s0 : (r < 256) ? s1 : (r < 384) ? s2 : s3;
  out[i] = src[(r & 127) * NGC + col];
}

// DCN_in[b, j] = stacked[2b + j/1024, j%1024] + baseline, stacked = [PCEs;PCIs]
// baseline = mean(MF spikes) + 0.3 (interleave does not change the mean)
__global__ void k_gather_dcn(const bf16_t* __restrict__ e, const bf16_t* __restrict__ ii,
                             const float* __restrict__ spikeSum, bf16_t* __restrict__ out)
{
  const int i = blockIdx.x * 256 + threadIdx.x;     // 128 * 2048 total
  if (i >= BATCH * 2 * NPC) return;
  const int b   = i >> 11;
  const int j   = i & 2047;
  const int seg = j >> 10;
  const int col = j & 1023;
  const int r   = 2 * b + seg;
  const bf16_t* src = (r < 128) ? e : ii;
  const float base = spikeSum[0] * (1.0f / (float)(BATCH * 4 * NMF)) + 0.3f;
  out[i] = (bf16_t)((float)src[(r & 127) * NPC + col] + base);
}

// ---------------- host orchestration ----------------

extern "C" void kernel_launch(void* const* d_in, const int* in_sizes, int n_in,
                              void* d_out, int out_size, void* d_ws, size_t ws_size,
                              hipStream_t stream) {
  (void)in_sizes; (void)n_in; (void)ws_size;

  // ---- input pointers (setup_inputs dict order) ----
  const float* x_f32[4]  = { (const float*)d_in[0], (const float*)d_in[1],
                             (const float*)d_in[2], (const float*)d_in[3] };
  const float* W_f32[11] = { (const float*)d_in[4],  (const float*)d_in[6],
                             (const float*)d_in[8],  (const float*)d_in[10],
                             (const float*)d_in[12], (const float*)d_in[14],
                             (const float*)d_in[16], (const float*)d_in[18],
                             (const float*)d_in[20], (const float*)d_in[22],
                             (const float*)d_in[24] };
  const float* bMF[4] = { (const float*)d_in[5],  (const float*)d_in[7],
                          (const float*)d_in[9],  (const float*)d_in[11] };
  const float* bGC[4] = { (const float*)d_in[13], (const float*)d_in[15],
                          (const float*)d_in[17], (const float*)d_in[19] };
  const float* bPC[2] = { (const float*)d_in[21], (const float*)d_in[23] };
  const float* bDCN   =   (const float*)d_in[25];

  // ---- workspace bump allocator (256B aligned) ----
  char* ws = (char*)d_ws;
  size_t off = 0;
  auto alloc = [&](size_t bytes) -> char* {
    char* p = ws + off;
    off += (bytes + 255) & ~(size_t)255;
    return p;
  };

  const size_t wElems[11] = {
    (size_t)NMF * KM, (size_t)NMF * KU, (size_t)NMF * KU, (size_t)NMF * KU,
    (size_t)NGC * NMF, (size_t)NGC * NMF, (size_t)NGC * NMF, (size_t)NGC * NMF,
    (size_t)NPC * 4 * NGC, (size_t)NPC * 4 * NGC,
    (size_t)NDCN * 2 * NPC };
  bf16_t* Wb[11];
  for (int i = 0; i < 11; ++i) Wb[i] = (bf16_t*)alloc(wElems[i] * sizeof(bf16_t));

  const size_t xElems[4] = { (size_t)BATCH * TSTEPS * KM, (size_t)BATCH * TSTEPS * KU,
                             (size_t)BATCH * TSTEPS * KU, (size_t)BATCH * TSTEPS * KU };
  bf16_t* Xb[4];
  for (int i = 0; i < 4; ++i) Xb[i] = (bf16_t*)alloc(xElems[i] * sizeof(bf16_t));

  // ---- persistent states (zeroed each call) ----
  const size_t stateStart = off;
  float  *memMF[4], *memGC[4], *memPC[2], *memDCN;
  bf16_t *spkMF[4], *spkGC[4], *spkPC[2], *spkDCN;
  for (int i = 0; i < 4; ++i) memMF[i] = (float*)alloc((size_t)BATCH * NMF * 4);
  for (int i = 0; i < 4; ++i) memGC[i] = (float*)alloc((size_t)BATCH * NGC * 4);
  for (int i = 0; i < 2; ++i) memPC[i] = (float*)alloc((size_t)BATCH * NPC * 4);
  memDCN = (float*)alloc((size_t)BATCH * NDCN * 4);
  for (int i = 0; i < 4; ++i) spkMF[i] = (bf16_t*)alloc((size_t)BATCH * NMF * 2);
  for (int i = 0; i < 4; ++i) spkGC[i] = (bf16_t*)alloc((size_t)BATCH * NGC * 2);
  for (int i = 0; i < 2; ++i) spkPC[i] = (bf16_t*)alloc((size_t)BATCH * NPC * 2);
  spkDCN = (bf16_t*)alloc((size_t)BATCH * NDCN * 2);
  float* spikeSum = (float*)alloc(sizeof(float));
  const size_t stateEnd = off;

  bf16_t* pfIn  = (bf16_t*)alloc((size_t)BATCH * 4 * NGC * 2);
  bf16_t* dcnIn = (bf16_t*)alloc((size_t)BATCH * 2 * NPC * 2);

  // ---- one-time (per call) conversions + state zeroing ----
  auto cvt = [&](const float* s, bf16_t* d, size_t n) {
    k_f32_to_bf16<<<(unsigned)((n + 255) / 256), 256, 0, stream>>>(s, d, n);
  };
  for (int i = 0; i < 11; ++i) cvt(W_f32[i], Wb[i], wElems[i]);
  for (int i = 0; i < 4;  ++i) cvt(x_f32[i], Xb[i], xElems[i]);
  {
    size_t nwords = (stateEnd - stateStart) / 4;
    k_zero_u32<<<512, 256, 0, stream>>>((unsigned*)(ws + stateStart), nwords);
  }

  auto gemm = [&](const bf16_t* X, int lda, int K, const bf16_t* W, const float* bias,
                  float* mem, bf16_t* spk, int N, float* ssum, float* accum) {
    k_gemm_mem_spike<<<dim3((unsigned)(N / 32)), dim3(256), 0, stream>>>(
        X, lda, K, W, bias, mem, spk, N, ssum, accum);
  };

  float* out = (float*)d_out;  // 128 x 1408 f32

  for (int outer = 0; outer < TSTEPS; ++outer) {
    // fresh DCN spike-sum accumulator for this outer step
    k_zero_u32<<<256, 256, 0, stream>>>((unsigned*)out, (size_t)out_size);

    for (int t = 0; t < TSTEPS; ++t) {
      k_zero_scalar<<<1, 1, 0, stream>>>(spikeSum);

      // MF layers: A is the time-t slice of the (B, T, D) input -> lda = T*D
      gemm(Xb[0] + (size_t)t * KM, TSTEPS * KM, KM, Wb[0], bMF[0],
           memMF[0], spkMF[0], NMF, spikeSum, nullptr);
      for (int i = 1; i < 4; ++i)
        gemm(Xb[i] + (size_t)t * KU, TSTEPS * KU, KU, Wb[i], bMF[i],
             memMF[i], spkMF[i], NMF, spikeSum, nullptr);

      // GC layers: A = MF spikes
      for (int i = 0; i < 4; ++i)
        gemm(spkMF[i], NMF, NMF, Wb[4 + i], bGC[i],
             memGC[i], spkGC[i], NGC, nullptr, nullptr);

      // PF_in interleave, then PCE / PCI
      k_gather_pf<<<(BATCH * 4 * NGC) / 256, 256, 0, stream>>>(
          spkGC[0], spkGC[1], spkGC[2], spkGC[3], pfIn);
      gemm(pfIn, 4 * NGC, 4 * NGC, Wb[8], bPC[0], memPC[0], spkPC[0], NPC, nullptr, nullptr);
      gemm(pfIn, 4 * NGC, 4 * NGC, Wb[9], bPC[1], memPC[1], spkPC[1], NPC, nullptr, nullptr);

      // DCN_in interleave (+ baseline from MF spike mean), then DCN
      k_gather_dcn<<<(BATCH * 2 * NPC) / 256, 256, 0, stream>>>(
          spkPC[0], spkPC[1], spikeSum, dcnIn);
      gemm(dcnIn, 2 * NPC, 2 * NPC, Wb[10], bDCN,
           memDCN, spkDCN, NDCN, nullptr, out);
    }
  }
  // after the last outer iteration, `out` holds sums[-1]
}
